// PatchedGNNLayer_56959856280340
// MI455X (gfx1250) — matrix operand order repaired and checked by reading, not compiled, not run on hardware
//
#include <hip/hip_runtime.h>
#include <hip/hip_bf16.h>
#include <math.h>

// ---------------------------------------------------------------------------
// PatchedGNNLayer for MI455X (gfx1250, wave32, WMMA).
// B=4, S=512, D=256.
//  - All GEMMs -> v_wmma_f32_16x16x32_bf16, 16x64 strip per wave (A reuse x4).
//  - Pairwise relu-score -> packed f16 VALU fed from LDS; hj chunks staged via
//    GLOBAL_LOAD_ASYNC_TO_LDS_B128 (ASYNCcnt) with 16-row i-tiling (16x less
//    L2 traffic than row-per-block).
//  - b2 dropped: softmax is invariant to a constant shift of the scores.
// ---------------------------------------------------------------------------

typedef __attribute__((ext_vector_type(16))) __bf16    v16bf;
typedef __attribute__((ext_vector_type(8)))  __bf16    v8bf;
typedef __attribute__((ext_vector_type(8)))  float     v8f;
typedef __attribute__((ext_vector_type(8)))  short     v8s;
typedef __attribute__((ext_vector_type(2)))  _Float16  h2;

#define NEGINF (-1e30f)

__device__ __forceinline__ unsigned short f2bf_bits(float f) {
    unsigned int u = __builtin_bit_cast(unsigned int, f);
    unsigned int r = u + 0x7FFFu + ((u >> 16) & 1u);   // RNE
    return (unsigned short)(r >> 16);
}

// --- WMMA fragment loaders (wave32 layouts, ISA 7.12.2) ---------------------
__device__ __forceinline__ v16bf frag_a(const __bf16* base, int row0, int ld,
                                        int k0, int lane) {
    int m  = lane & 15;
    int kh = (lane >> 4) << 3;                  // 0 or 8
    const __bf16* p = base + (size_t)(row0 + m) * ld + k0 + kh;
    v8bf lo = *(const v8bf*)p;                  // 16B aligned
    v8bf hi = *(const v8bf*)(p + 16);
    return __builtin_shufflevector(lo, hi, 0, 1, 2, 3, 4, 5, 6, 7,
                                           8, 9, 10, 11, 12, 13, 14, 15);
}
__device__ __forceinline__ v16bf frag_b(const __bf16* baseT, int n0, int ld,
                                        int k0, int lane) {
    int n    = lane & 15;
    int koff = ((lane >> 4) & 1) << 3;          // 0 or 8
    const __bf16* p = baseT + (size_t)(n0 + n) * ld + k0 + koff;
    v8bf lo = *(const v8bf*)p;
    v8bf hi = *(const v8bf*)(p + 16);
    return __builtin_shufflevector(lo, hi, 0, 1, 2, 3, 4, 5, 6, 7,
                                           8, 9, 10, 11, 12, 13, 14, 15);
}

// --- prep: x -> bf16 --------------------------------------------------------
__global__ void k_prep_x(const float* __restrict__ x, __bf16* __restrict__ xbf) {
    int i = blockIdx.x * 256 + threadIdx.x;
    xbf[i] = (__bf16)x[i];
}

// --- prep: fused weight WcatT (768 x 256) and WoT (256 x 256), N-major ------
__global__ void k_prep_w(const float* __restrict__ W1, const float* __restrict__ Wv,
                         const float* __restrict__ Wo,
                         __bf16* __restrict__ WcatT, __bf16* __restrict__ WoT) {
    int i = blockIdx.x * 256 + threadIdx.x;
    int n = i >> 8, k = i & 255;
    float v;
    if (n < 256)      v = W1[k * 256 + n] + W1[(512 + k) * 256 + n];       // Wa+Wc
    else if (n < 512) v = W1[(256 + k) * 256 + (n - 256)]
                        - W1[(512 + k) * 256 + (n - 256)];                  // Wb-Wc
    else              v = Wv[k * 256 + (n - 512)];                          // Wv
    WcatT[(size_t)n * 256 + k] = (__bf16)v;
    if (i < 256 * 256) {
        int e = i >> 8, d = i & 255;
        WoT[(size_t)e * 256 + d] = (__bf16)Wo[d * 256 + e];
    }
}

// --- GEMM 1: H = x @ Wcat, 16x64 strip per wave -----------------------------
// HT layout (per batch, 16-bit slots):
//   region 0 [n=0..255]  : hi'(+b1) f16, transposed [h][s]
//   region 1 [n=256..511]: hj f16, h-pair interleaved: slot (h/2)*1024+s*2+(h&1)
//   region 2 [n=512..767]: values(+bv) bf16, transposed [d][s]
__global__ void k_gemm_qkv(const __bf16* __restrict__ xbf,
                           const __bf16* __restrict__ WcatT,
                           const float* __restrict__ b1,
                           const float* __restrict__ bv,
                           unsigned short* __restrict__ HT) {
    int lane = threadIdx.x & 31, w = threadIdx.x >> 5;
    int t = blockIdx.x * 8 + w;                  // 0..1535 = 128 x 12 strips
    int g = t % 12, tileM = t / 12;
    int n0 = g * 64, row0 = tileM * 16;

    v8f c[4] = {{}, {}, {}, {}};
#pragma unroll
    for (int k0 = 0; k0 < 256; k0 += 32) {
        if (k0 + 32 < 256)
            __builtin_prefetch(xbf + (size_t)(row0 + (lane & 15)) * 256 + k0 + 32, 0, 1);
        v16bf a = frag_a(xbf, row0, 256, k0, lane);
#pragma unroll
        for (int q = 0; q < 4; ++q) {
            v16bf bb = frag_b(WcatT, n0 + q * 16, 256, k0, lane);
            c[q] = __builtin_amdgcn_wmma_f32_16x16x32_bf16(false, a, false, bb,
                                                           (short)0, c[q], false, false);
        }
    }

    int b = row0 >> 9, s_in = row0 & 511;
    int r8 = ((lane >> 4) & 1) << 3;
    int region = n0 >> 8;                        // uniform per 64-wide strip
    _Float16* HT16 = (_Float16*)HT;
#pragma unroll
    for (int q = 0; q < 4; ++q) {
        int n = n0 + q * 16 + (lane & 15);
        if (region == 1) {                       // hj: h-pair interleaved
            int h = n - 256, p = h >> 1, ph = h & 1;
            size_t base = (size_t)(b * 768 + 256) * 512 + (size_t)p * 1024 + ph;
#pragma unroll
            for (int r = 0; r < 8; ++r)
                HT16[base + (size_t)(s_in + r8 + r) * 2] = (_Float16)c[q][r];
        } else {
            float bias = (region == 0) ? b1[n] : bv[n - 512];
            v8s o;
#pragma unroll
            for (int r = 0; r < 8; ++r) {
                float v = c[q][r] + bias;
                unsigned short us = (region == 0)
                    ? __builtin_bit_cast(unsigned short, (_Float16)v)
                    : f2bf_bits(v);
                o[r] = (short)us;
            }
            *(v8s*)(HT + ((size_t)(b * 768 + n)) * 512 + s_in + r8) = o;
        }
    }
}

// --- scores + masked softmax ------------------------------------------------
// One 512-thread block per (b, 16-row i-tile).  Wave ww owns row i0+ww.
// hj staged in 64KB LDS chunks via GLOBAL_LOAD_ASYNC_TO_LDS_B128 (ASYNCcnt).
__global__ void k_scores(const unsigned short* __restrict__ HT,
                         const float* __restrict__ w2,
                         unsigned short* __restrict__ attn) {
    extern __shared__ char smem[];
    _Float16* hjL = (_Float16*)smem;             // 65536 B: [128 hpairs][128 j][2]
    _Float16* hiL = (_Float16*)(smem + 65536);   //  8192 B: [16 ii][256 h]
    _Float16* w2L = (_Float16*)(smem + 73728);   //   512 B

    int b = blockIdx.y, i0 = blockIdx.x * 16, tid = threadIdx.x;
    int ww = tid >> 5, lane = tid & 31;
    int i = i0 + ww;

    // stage hi rows (strided column gather, once per block) and w2
#pragma unroll
    for (int it = 0; it < 8; ++it) {
        int L = it * 512 + tid;
        int h = L >> 4, ii = L & 15;
        hiL[ii * 256 + h] =
            ((const _Float16*)HT)[((size_t)(b * 768 + h)) * 512 + i0 + ii];
    }
    if (tid < 256) w2L[tid] = (_Float16)w2[tid];
    __syncthreads();

    const char* hjGlob = (const char*)(HT + (size_t)(b * 768 + 256) * 512);
    unsigned ldsBase = (unsigned)(size_t)(void*)hjL;   // raw LDS byte offset
    float sc[16];
    h2 zero = (h2)(_Float16)0;

    for (int cch = 0; cch < 4; ++cch) {
        int j0 = cch * 128;
        // async stage: 128 hpair-rows x 512B -> LDS (4096 b128, 8 per thread)
#pragma unroll
        for (int it = 0; it < 8; ++it) {
            int L = it * 512 + tid;
            int p = L >> 5, o = L & 31;
            unsigned lds = ldsBase + (unsigned)(p * 512 + o * 16);
            unsigned long long ga = (unsigned long long)(size_t)hjGlob
                                  + (size_t)p * 2048 + (size_t)j0 * 4 + (size_t)o * 16;
            asm volatile("global_load_async_to_lds_b128 %0, %1, off"
                         :: "v"(lds), "v"(ga) : "memory");
        }
        asm volatile("s_wait_asynccnt 0x0" ::: "memory");
        __syncthreads();

        h2 acc[4] = {zero, zero, zero, zero};
        const h2* hi2 = (const h2*)(hiL + ww * 256);
        const h2* w22 = (const h2*)w2L;
#pragma unroll 4
        for (int p = 0; p < 128; ++p) {
            h2 hv = hi2[p];
            h2 wv = w22[p];
            const _Float16* rowp = hjL + p * 256;
#pragma unroll
            for (int q = 0; q < 4; ++q) {
                h2 jv = *(const h2*)(rowp + (lane + 32 * q) * 2);
                h2 tv = hv + jv;                              // v_pk_add_f16
                tv = __builtin_elementwise_max(tv, zero);     // v_pk_max (ReLU)
                acc[q] += tv * wv;                            // v_pk_fma_f16
            }
        }
#pragma unroll
        for (int q = 0; q < 4; ++q) {
            int j = j0 + q * 32 + lane;
            sc[cch * 4 + q] = (j < i) ? ((float)acc[q].x + (float)acc[q].y)
                                      : NEGINF;
        }
        __syncthreads();                          // before hjL is overwritten
    }

    // row softmax entirely within wave ww (row's 512 scores live in this wave)
    float mx = sc[0];
#pragma unroll
    for (int k = 1; k < 16; ++k) mx = fmaxf(mx, sc[k]);
#pragma unroll
    for (int off = 16; off > 0; off >>= 1) mx = fmaxf(mx, __shfl_xor(mx, off, 32));
    float pv[16], tot = 0.f;
#pragma unroll
    for (int k = 0; k < 16; ++k) { pv[k] = __expf(sc[k] - mx); tot += pv[k]; }
#pragma unroll
    for (int off = 16; off > 0; off >>= 1) tot += __shfl_xor(tot, off, 32);
    float inv = (i > 0 && tot > 0.f) ? (1.f / tot) : 0.f;   // row 0 -> zeros
#pragma unroll
    for (int k = 0; k < 16; ++k) {
        int j = (k >> 2) * 128 + (k & 3) * 32 + lane;
        attn[((size_t)(b * 512 + i)) * 512 + j] = f2bf_bits(pv[k] * inv);
    }
}

// --- GEMM 2: message = attn @ values, 16x64 strip per wave ------------------
__global__ void k_message(const unsigned short* __restrict__ attn_us,
                          const unsigned short* __restrict__ HT,
                          unsigned short* __restrict__ msg) {
    int lane = threadIdx.x & 31, w = threadIdx.x >> 5;
    int t = blockIdx.x * 8 + w;                   // 0..511
    int b = t >> 7, tt = t & 127;
    int tileM = tt >> 2, g = tt & 3;
    int row0 = b * 512 + tileM * 16;              // global attn row
    int n0 = g * 64;

    const __bf16* attn = (const __bf16*)attn_us;
    const __bf16* vT   = (const __bf16*)HT + ((size_t)(b * 768 + 512)) * 512;

    v8f c[4] = {{}, {}, {}, {}};
#pragma unroll 4
    for (int k0 = 0; k0 < 512; k0 += 32) {
        v16bf a = frag_a(attn, row0, 512, k0, lane);
#pragma unroll
        for (int q = 0; q < 4; ++q) {
            v16bf bb = frag_b(vT, n0 + q * 16, 512, k0, lane);
            c[q] = __builtin_amdgcn_wmma_f32_16x16x32_bf16(false, a, false, bb,
                                                           (short)0, c[q], false, false);
        }
    }
    int r8 = ((lane >> 4) & 1) << 3;
#pragma unroll
    for (int q = 0; q < 4; ++q) {
        int d = n0 + q * 16 + (lane & 15);
#pragma unroll
        for (int r = 0; r < 8; ++r) {
            int ir = row0 + r + r8;
            msg[(size_t)ir * 256 + d] = f2bf_bits(c[q][r]);
        }
    }
}

// --- GEMM 3 + residual + LayerNorm: 512 threads = 16 waves cover 16x256 -----
__global__ void k_out_ln(const unsigned short* __restrict__ msg_us,
                         const __bf16* __restrict__ WoT,
                         const float* __restrict__ x,
                         const float* __restrict__ bo,
                         const float* __restrict__ gamma,
                         const float* __restrict__ beta,
                         float* __restrict__ out) {
    __shared__ float sums[16], sqs[16];
    int tid = threadIdx.x, lane = tid & 31, w = tid >> 5;
    if (tid < 16) { sums[tid] = 0.f; sqs[tid] = 0.f; }
    __syncthreads();

    int row0 = blockIdx.x * 16;
    int n0 = w * 16;
    const __bf16* msgb = (const __bf16*)msg_us;

    v8f c = {};
#pragma unroll
    for (int k0 = 0; k0 < 256; k0 += 32) {
        v16bf a  = frag_a(msgb, row0, 256, k0, lane);
        v16bf bb = frag_b(WoT,  n0,   256, k0, lane);
        c = __builtin_amdgcn_wmma_f32_16x16x32_bf16(false, a, false, bb,
                                                    (short)0, c, false, false);
    }

    int n  = n0 + (lane & 15);
    int r8 = ((lane >> 4) & 1) << 3;
    float vals[8];
#pragma unroll
    for (int r = 0; r < 8; ++r) {
        int row = row0 + r + r8;
        float v = c[r] + bo[n] + x[(size_t)row * 256 + n];   // residual
        vals[r] = v;
        atomicAdd(&sums[r + r8], v);              // ds_add_f32
        atomicAdd(&sqs[r + r8], v * v);
    }
    __syncthreads();
#pragma unroll
    for (int r = 0; r < 8; ++r) {
        int lr = r + r8, row = row0 + lr;
        float mu  = sums[lr] * (1.f / 256.f);
        float var = sqs[lr] * (1.f / 256.f) - mu * mu;
        out[(size_t)row * 256 + n] =
            (vals[r] - mu) * rsqrtf(var + 1e-5f) * gamma[n] + beta[n];
    }
}

// ---------------------------------------------------------------------------
extern "C" void kernel_launch(void* const* d_in, const int* in_sizes, int n_in,
                              void* d_out, int out_size, void* d_ws, size_t ws_size,
                              hipStream_t stream) {
    const float* x     = (const float*)d_in[0];
    const float* W1    = (const float*)d_in[1];
    const float* b1    = (const float*)d_in[2];
    const float* w2    = (const float*)d_in[3];
    /* d_in[4] = b2: constant shift of scores, cancels in softmax */
    const float* Wv    = (const float*)d_in[5];
    const float* bv    = (const float*)d_in[6];
    const float* Wo    = (const float*)d_in[7];
    const float* bo    = (const float*)d_in[8];
    const float* gamma = (const float*)d_in[9];
    const float* beta  = (const float*)d_in[10];
    (void)in_sizes; (void)n_in; (void)out_size; (void)ws_size;

    char* ws = (char*)d_ws;                       // ~7.9MB, L2-resident
    __bf16*         xbf   = (__bf16*)(ws + 0);                 // 1 MB
    __bf16*         WcatT = (__bf16*)(ws + 1048576);           // 384 KB
    __bf16*         WoT   = (__bf16*)(ws + 1441792);           // 128 KB
    unsigned short* HT    = (unsigned short*)(ws + 1572864);   // 3 MB
    unsigned short* attn  = (unsigned short*)(ws + 4718592);   // 2 MB
    unsigned short* msg   = (unsigned short*)(ws + 6815744);   // 1 MB

    k_prep_x  <<<2048, 256, 0, stream>>>(x, xbf);
    k_prep_w  <<<768,  256, 0, stream>>>(W1, Wv, Wo, WcatT, WoT);
    k_gemm_qkv<<<192,  256, 0, stream>>>(xbf, WcatT, b1, bv, HT);
    k_scores  <<<dim3(32, 4), 512, 74240, stream>>>(HT, w2, attn);
    k_message <<<64,   256, 0, stream>>>(attn, HT, msg);
    k_out_ln  <<<128,  512, 0, stream>>>(msg, WoT, x, bo, gamma, beta,
                                         (float*)d_out);
}
